// SymmetricLinearAttention_46084999086237
// MI455X (gfx1250) — matrix operand order repaired and checked by reading, not compiled, not run on hardware
//
#include <hip/hip_runtime.h>
#include <hip/hip_bf16.h>

// ---------------------------------------------------------------------------
// SymmetricLinearAttention for MI455X (gfx1250): bf16 WMMA pipeline with
// async global->LDS staging of B tiles (double buffered) in the big GEMMs.
// Per-wave tile 32x64: every LDS B fragment feeds two WMMAs.
// S=2048 B=4 D=1024 H=16 HD=64.
// ---------------------------------------------------------------------------

#define S_LEN 2048
#define BATCH 4
#define DMODEL 1024
#define NHEAD 16
#define HDIM 64
#define SB (S_LEN * BATCH)   // 8192 GEMM rows
#define EPS 1e-6f

typedef __bf16 bf16_t;
typedef __attribute__((ext_vector_type(8)))  __bf16 v8bf;
typedef __attribute__((ext_vector_type(16))) __bf16 v16bf;
typedef __attribute__((ext_vector_type(8)))  float  v8f;

// ---- WMMA fragment loaders (contiguous k = 16*(lane/16) + s, same order in A and B) ----------
__device__ __forceinline__ v16bf load_A_tile(const bf16_t* __restrict__ p,
                                             int ld, int m0, int k0, int lane) {
  const int row = m0 + (lane & 15);
  const int c   = k0 + ((lane >> 4) << 4);
  const bf16_t* q = p + (size_t)row * ld + c;
  v8bf lo = *reinterpret_cast<const v8bf*>(q);
  v8bf hi = *reinterpret_cast<const v8bf*>(q + 8);
  v16bf r;
#pragma unroll
  for (int i = 0; i < 8; ++i) { r[i] = lo[i]; r[i + 8] = hi[i]; }
  return r;
}

__device__ __forceinline__ v16bf load_B_tile(const bf16_t* __restrict__ p,
                                             int ld, int n0, int k0, int lane) {
  const int row = n0 + (lane & 15);
  const int c   = k0 + ((lane >> 4) << 4);
  const bf16_t* q = p + (size_t)row * ld + c;
  v8bf lo = *reinterpret_cast<const v8bf*>(q);
  v8bf hi = *reinterpret_cast<const v8bf*>(q + 8);
  v16bf r;
#pragma unroll
  for (int i = 0; i < 8; ++i) { r[i] = lo[i]; r[i + 8] = hi[i]; }
  return r;
}

// B fragment from an LDS-resident 64(N) x 32(K) tile, row-major [64][32].
__device__ __forceinline__ v16bf frag_B_lds(const bf16_t* tile, int j, int lane) {
  const bf16_t* q = tile + (j * 16 + (lane & 15)) * 32 + ((lane >> 4) << 4);
  v8bf lo = *reinterpret_cast<const v8bf*>(q);
  v8bf hi = *reinterpret_cast<const v8bf*>(q + 8);
  v16bf r;
#pragma unroll
  for (int i = 0; i < 8; ++i) { r[i] = lo[i]; r[i + 8] = hi[i]; }
  return r;
}

__device__ __forceinline__ v8f wmma_bf16(v16bf a, v16bf b, v8f c) {
  return __builtin_amdgcn_wmma_f32_16x16x32_bf16(
      false, a, false, b, (short)0, c, false, false);
}

// Async global->LDS copy (ASYNCcnt-tracked; VGLOBAL "GV" addressing).
__device__ __forceinline__ void async_copy_b128(const bf16_t* g, bf16_t* l) {
  unsigned lds_addr = (unsigned)(size_t)l;   // LDS aperture: low 32 bits = LDS addr
  asm volatile("global_load_async_to_lds_b128 %0, %1, off"
               :: "v"(lds_addr), "v"(g) : "memory");
}
__device__ __forceinline__ void wait_async0() {
  asm volatile("s_wait_asynccnt 0x0" ::: "memory");
}

// Async fill of one 64x32 bf16 B tile (rows n0.., cols k0..k0+31, row-major
// [*, ld]) into LDS buf[64][32]. 128 threads, 2 x b128 segments each.
__device__ __forceinline__ void fill_B_async(const bf16_t* __restrict__ W, int ld,
                                             int n0, int k0, bf16_t* buf, int tid) {
#pragma unroll
  for (int i = 0; i < 2; ++i) {
    const int slot = tid + 128 * i;        // 0..255
    const int row  = slot >> 2;            // 0..63
    const int seg  = (slot & 3) << 3;      // 0,8,16,24 halves
    async_copy_b128(W + (size_t)(n0 + row) * ld + k0 + seg,
                    buf + row * 32 + seg);
  }
}

__device__ __forceinline__ float phi_elu1(float v) {
  return v > 0.f ? v + 1.f : __expf(v);
}

// ---- Kernel 0a: f32 -> bf16 elementwise -----------------------------------
__global__ __launch_bounds__(256) void cvt_bf16(const float* __restrict__ src,
                                                bf16_t* __restrict__ dst, int n) {
  int i = blockIdx.x * 256 + threadIdx.x;
  if (i < n) dst[i] = (bf16_t)src[i];
}

// ---- Kernel 0b: f32 -> bf16 transpose (Wq -> Wq^T for final projection B) -
__global__ __launch_bounds__(256) void cvt_bf16_T(const float* __restrict__ src,
                                                  bf16_t* __restrict__ dst) {
  int i = blockIdx.x * 256 + threadIdx.x;
  if (i < DMODEL * DMODEL) {
    int d = i >> 10, e = i & 1023;
    dst[(size_t)e * DMODEL + d] = (bf16_t)src[i];
  }
}

// ---- Kernel 1: projection GEMM + bias + phi, dual output layouts ----------
// Block tile 128(M) x 64(N); wave tile 32 x 64 (2 A frags x 4 B frags).
// mode 0 (Q): out = qf  [B,H,S,64]; mode 1 (K): out = kf^T [B,H,64,S]
__global__ __launch_bounds__(128) void gemm_proj_phi(
    const bf16_t* __restrict__ A, const bf16_t* __restrict__ W,
    const float* __restrict__ bias, bf16_t* __restrict__ out, int mode) {
  __shared__ bf16_t smemB[2][64 * 32];     // double-buffered B tile (2 x 4 KB)
  const int tid  = threadIdx.x;
  const int lane = tid & 31;
  const int wave = tid >> 5;
  const int m0 = blockIdx.x * 128 + wave * 32;
  const int n0 = blockIdx.y * 64;

  fill_B_async(W, DMODEL, n0, 0, smemB[0], tid);
  v16bf a0 = load_A_tile(A, DMODEL, m0,      0, lane);
  v16bf a1 = load_A_tile(A, DMODEL, m0 + 16, 0, lane);
  wait_async0();
  __syncthreads();

  v8f acc[8] = {};
  int p = 0;
  for (int k0 = 0; k0 < DMODEL; k0 += 32) {
    v16bf a0n, a1n;
    if (k0 + 32 < DMODEL) {                 // overlap next fill with WMMA
      fill_B_async(W, DMODEL, n0, k0 + 32, smemB[p ^ 1], tid);
      a0n = load_A_tile(A, DMODEL, m0,      k0 + 32, lane);
      a1n = load_A_tile(A, DMODEL, m0 + 16, k0 + 32, lane);
    }
    v16bf b0 = frag_B_lds(smemB[p], 0, lane);
    v16bf b1 = frag_B_lds(smemB[p], 1, lane);
    v16bf b2 = frag_B_lds(smemB[p], 2, lane);
    v16bf b3 = frag_B_lds(smemB[p], 3, lane);
    acc[0] = wmma_bf16(a0, b0, acc[0]);  acc[4] = wmma_bf16(a1, b0, acc[4]);
    acc[1] = wmma_bf16(a0, b1, acc[1]);  acc[5] = wmma_bf16(a1, b1, acc[5]);
    acc[2] = wmma_bf16(a0, b2, acc[2]);  acc[6] = wmma_bf16(a1, b2, acc[6]);
    acc[3] = wmma_bf16(a0, b3, acc[3]);  acc[7] = wmma_bf16(a1, b3, acc[7]);
    a0 = a0n; a1 = a1n;
    wait_async0();
    __syncthreads();
    p ^= 1;
  }

  const int col    = lane & 15;
  const int rowOff = (lane >> 4) * 8;
#pragma unroll
  for (int mi = 0; mi < 2; ++mi)
#pragma unroll
    for (int j = 0; j < 4; ++j) {
      const int n = n0 + j * 16 + col;
      const float bv = bias[n];
      const int h = n >> 6, hd = n & 63;
#pragma unroll
      for (int v = 0; v < 8; ++v) {
        const int r = m0 + mi * 16 + rowOff + v;   // r = s*B + b
        const int s = r >> 2, b = r & 3;
        const bf16_t o = (bf16_t)phi_elu1(acc[mi * 4 + j][v] + bv);
        if (mode == 0)
          out[((size_t)(b * NHEAD + h) * S_LEN + s) * HDIM + hd] = o;
        else
          out[((size_t)(b * NHEAD + h) * HDIM + hd) * S_LEN + s] = o;
      }
    }
}

// ---- Kernel 2: per-head kv = kf^T kf (64x64, K=2048) + ksum via ones-tile -
__global__ __launch_bounds__(128) void head_kv(
    const bf16_t* __restrict__ kft, bf16_t* __restrict__ kv,
    float* __restrict__ ksum) {
  const int lane = threadIdx.x & 31;
  const int wave = threadIdx.x >> 5;
  const int bh = blockIdx.x;
  const bf16_t* base = kft + (size_t)bh * HDIM * S_LEN;
  const int d0 = wave * 16;

  v16bf ones;
#pragma unroll
  for (int i = 0; i < 16; ++i) ones[i] = (bf16_t)1.0f;

  v8f acc[4] = {}; v8f accs = {};
  for (int s0 = 0; s0 < S_LEN; s0 += 32) {
    if (s0 + 32 < S_LEN)
      __builtin_prefetch(base + (size_t)(d0 + (lane & 15)) * S_LEN + s0 + 32, 0, 3);
    v16bf a = load_A_tile(base, S_LEN, d0, s0, lane);
#pragma unroll
    for (int j = 0; j < 4; ++j) {
      v16bf b = load_B_tile(base, S_LEN, j * 16, s0, lane);
      acc[j] = wmma_bf16(a, b, acc[j]);
    }
    accs = wmma_bf16(a, ones, accs);        // columns all equal sum_s kf
  }

  const int col    = lane & 15;
  const int rowOff = (lane >> 4) * 8;
#pragma unroll
  for (int j = 0; j < 4; ++j)
#pragma unroll
    for (int v = 0; v < 8; ++v) {
      const int d = d0 + rowOff + v;
      kv[(size_t)bh * 4096 + d * 64 + j * 16 + col] = (bf16_t)acc[j][v];
    }
  if (col == 0) {
#pragma unroll
    for (int v = 0; v < 8; ++v)
      ksum[bh * 64 + d0 + rowOff + v] = accs[v];
  }
}

// ---- Kernel 3: attn = (qf @ kv) / (qf . ksum + eps); z via broadcast tile -
__global__ __launch_bounds__(128) void head_attn(
    const bf16_t* __restrict__ qf, const bf16_t* __restrict__ kv,
    const float* __restrict__ ksum, bf16_t* __restrict__ attn) {
  const int lane = threadIdx.x & 31;
  const int wave = threadIdx.x >> 5;
  const int bh = blockIdx.y;
  const bf16_t* qb  = qf + (size_t)bh * S_LEN * HDIM;
  const bf16_t* kvb = kv + (size_t)bh * 4096;
  const float*  ks  = ksum + bh * 64;
  const int m0 = blockIdx.x * 64 + wave * 16;

  v8f acc[4] = {}; v8f zacc = {};
#pragma unroll
  for (int k0 = 0; k0 < HDIM; k0 += 32) {
    v16bf a = load_A_tile(qb, HDIM, m0, k0, lane);
#pragma unroll
    for (int j = 0; j < 4; ++j) {
      v16bf b = load_B_tile(kvb, HDIM, j * 16, k0, lane);   // kv symmetric
      acc[j] = wmma_bf16(a, b, acc[j]);
    }
    v16bf bk;                                // B2[k,n] = ksum[k] for all n
    const int c = k0 + ((lane >> 4) << 4);
#pragma unroll
    for (int i = 0; i < 16; ++i) bk[i] = (bf16_t)ks[c + i];
    zacc = wmma_bf16(a, bk, zacc);           // each elem = z[row]
  }

  const int b_ = bh >> 4, h = bh & 15;
  const int col    = lane & 15;
  const int rowOff = (lane >> 4) * 8;
#pragma unroll
  for (int j = 0; j < 4; ++j)
#pragma unroll
    for (int v = 0; v < 8; ++v) {
      const int sg = m0 + rowOff + v;
      const float val = acc[j][v] / (zacc[v] + EPS);
      const size_t r = (size_t)sg * BATCH + b_;
      attn[r * DMODEL + h * 64 + j * 16 + col] = (bf16_t)val;
    }
}

// ---- Kernel 4: final projection y = attn @ Wq (B operand = Wq^T rows) -----
__global__ __launch_bounds__(128) void gemm_out(
    const bf16_t* __restrict__ A, const bf16_t* __restrict__ WT,
    float* __restrict__ out) {
  __shared__ bf16_t smemB[2][64 * 32];
  const int tid  = threadIdx.x;
  const int lane = tid & 31;
  const int wave = tid >> 5;
  const int m0 = blockIdx.x * 128 + wave * 32;
  const int n0 = blockIdx.y * 64;

  fill_B_async(WT, DMODEL, n0, 0, smemB[0], tid);
  v16bf a0 = load_A_tile(A, DMODEL, m0,      0, lane);
  v16bf a1 = load_A_tile(A, DMODEL, m0 + 16, 0, lane);
  wait_async0();
  __syncthreads();

  v8f acc[8] = {};
  int p = 0;
  for (int k0 = 0; k0 < DMODEL; k0 += 32) {
    v16bf a0n, a1n;
    if (k0 + 32 < DMODEL) {
      fill_B_async(WT, DMODEL, n0, k0 + 32, smemB[p ^ 1], tid);
      a0n = load_A_tile(A, DMODEL, m0,      k0 + 32, lane);
      a1n = load_A_tile(A, DMODEL, m0 + 16, k0 + 32, lane);
    }
    v16bf b0 = frag_B_lds(smemB[p], 0, lane);
    v16bf b1 = frag_B_lds(smemB[p], 1, lane);
    v16bf b2 = frag_B_lds(smemB[p], 2, lane);
    v16bf b3 = frag_B_lds(smemB[p], 3, lane);
    acc[0] = wmma_bf16(a0, b0, acc[0]);  acc[4] = wmma_bf16(a1, b0, acc[4]);
    acc[1] = wmma_bf16(a0, b1, acc[1]);  acc[5] = wmma_bf16(a1, b1, acc[5]);
    acc[2] = wmma_bf16(a0, b2, acc[2]);  acc[6] = wmma_bf16(a1, b2, acc[6]);
    acc[3] = wmma_bf16(a0, b3, acc[3]);  acc[7] = wmma_bf16(a1, b3, acc[7]);
    a0 = a0n; a1 = a1n;
    wait_async0();
    __syncthreads();
    p ^= 1;
  }

  const int col    = lane & 15;
  const int rowOff = (lane >> 4) * 8;
#pragma unroll
  for (int mi = 0; mi < 2; ++mi)
#pragma unroll
    for (int j = 0; j < 4; ++j)
#pragma unroll
      for (int v = 0; v < 8; ++v) {
        const size_t r = (size_t)(m0 + mi * 16 + rowOff + v);
        out[r * DMODEL + n0 + j * 16 + col] = acc[mi * 4 + j][v];
      }
}

// ---------------------------------------------------------------------------
extern "C" void kernel_launch(void* const* d_in, const int* in_sizes, int n_in,
                              void* d_out, int out_size, void* d_ws, size_t ws_size,
                              hipStream_t stream) {
  const float* h   = (const float*)d_in[0];
  const float* x   = (const float*)d_in[1];
  const float* wq  = (const float*)d_in[2];
  const float* wqb = (const float*)d_in[3];
  const float* wk  = (const float*)d_in[4];
  const float* wkb = (const float*)d_in[5];
  float* out = (float*)d_out;

  char* ws = (char*)d_ws;
  size_t off = 0;
  auto alloc = [&](size_t bytes) { void* p = ws + off; off = (off + bytes + 255) & ~(size_t)255; return p; };

  bf16_t* h_bf   = (bf16_t*)alloc((size_t)SB * DMODEL * 2);
  bf16_t* x_bf   = (bf16_t*)alloc((size_t)SB * DMODEL * 2);
  bf16_t* wq_bf  = (bf16_t*)alloc((size_t)DMODEL * DMODEL * 2);
  bf16_t* wk_bf  = (bf16_t*)alloc((size_t)DMODEL * DMODEL * 2);
  bf16_t* wqT_bf = (bf16_t*)alloc((size_t)DMODEL * DMODEL * 2);
  bf16_t* qf_bf  = (bf16_t*)alloc((size_t)SB * DMODEL * 2);
  bf16_t* kft_bf = (bf16_t*)alloc((size_t)SB * DMODEL * 2);
  bf16_t* kv_bf  = (bf16_t*)alloc((size_t)BATCH * NHEAD * 64 * 64 * 2);
  float*  ksum   = (float*) alloc((size_t)BATCH * NHEAD * 64 * 4);
  bf16_t* at_bf  = (bf16_t*)alloc((size_t)SB * DMODEL * 2);
  (void)ws_size; (void)n_in; (void)in_sizes; (void)out_size;

  const int nHX = SB * DMODEL;
  const int nW  = DMODEL * DMODEL;

  cvt_bf16<<<(nHX + 255) / 256, 256, 0, stream>>>(h, h_bf, nHX);
  cvt_bf16<<<(nHX + 255) / 256, 256, 0, stream>>>(x, x_bf, nHX);
  cvt_bf16<<<(nW + 255) / 256, 256, 0, stream>>>(wq, wq_bf, nW);
  cvt_bf16<<<(nW + 255) / 256, 256, 0, stream>>>(wk, wk_bf, nW);
  cvt_bf16_T<<<(nW + 255) / 256, 256, 0, stream>>>(wq, wqT_bf);

  dim3 gGemm(SB / 128, DMODEL / 64);   // 64 x 16 blocks, 4 waves each
  gemm_proj_phi<<<gGemm, 128, 0, stream>>>(h_bf, wq_bf, wqb, qf_bf, 0);
  gemm_proj_phi<<<gGemm, 128, 0, stream>>>(x_bf, wk_bf, wkb, kft_bf, 1);
  head_kv<<<BATCH * NHEAD, 128, 0, stream>>>(kft_bf, kv_bf, ksum);
  head_attn<<<dim3(S_LEN / 64, BATCH * NHEAD), 128, 0, stream>>>(qf_bf, kv_bf, ksum, at_bf);
  gemm_out<<<SB / 128 == 0 ? dim3(1,1) : dim3(SB / 128, DMODEL / 64), 128, 0, stream>>>(at_bf, wqT_bf, out);
}
